// GraphDegreeConv_56934086476262
// MI455X (gfx1250) — compile-verified, matrix operand order checked by reading, not run on hardware
//
#include <hip/hip_runtime.h>
#include <hip/hip_bf16.h>

// ---------------------------------------------------------------------------
// GraphDegreeConv fused kernel set for MI455X (gfx1250, wave32, WMMA)
//
// x[i,:] = W_self @ node[i] + W_deg_d @ concat(sum_nbr_node, sum_nbr_edge) + bias
// then BatchNorm1d (batch stats) + ReLU.
//
// Unified as A[16 x 320] (bf16, LDS) x Wc_d[128 x 320] (bf16, L2-resident)
// -> 16x128 f32 tile via v_wmma_f32_16x16x32_bf16, 10 K-chunks of 32.
// ---------------------------------------------------------------------------

typedef __attribute__((ext_vector_type(16))) __bf16 v16bf;
typedef __attribute__((ext_vector_type(8)))  __bf16 v8bf;
typedef __attribute__((ext_vector_type(8)))  float  v8f;

#define NNODE   400000
#define NEDGE   1000000
#define FNODE   128
#define FEDGE   64
#define FOUT    128
#define NPD     100000          // nodes per degree bucket
#define KTOT    320             // 128 self + 128 nbr-node + 64 nbr-edge
#define TPB     (NPD / 16)      // 16-row tiles per bucket = 6250
#define NTILES  (4 * TPB)       // 25000
#define LDS_STRIDE 336          // 320 + 16 pad (keeps 16B alignment of chunks)

// ---------------------------------------------------------------------------
// Kernel 0: build combined bf16 weights Wc[d][n][k], k-major rows of 320.
//   k <  128 : W_self[n][k]
//   k >= 128 : W_deg_d[n][k-128]   (W_deg is [128 x 192], node part then edge)
// ---------------------------------------------------------------------------
__global__ void gdc_prep_weights(const float* __restrict__ Wself,
                                 const float* __restrict__ Wd1,
                                 const float* __restrict__ Wd2,
                                 const float* __restrict__ Wd3,
                                 const float* __restrict__ Wd4,
                                 __bf16* __restrict__ Wc) {
    int e = blockIdx.x * blockDim.x + threadIdx.x;
    const int total = 4 * FOUT * KTOT;
    if (e >= total) return;
    int k = e % KTOT;
    int n = (e / KTOT) % FOUT;
    int d = e / (KTOT * FOUT);
    const float* Wd = (d == 0) ? Wd1 : (d == 1) ? Wd2 : (d == 2) ? Wd3 : Wd4;
    float v = (k < FNODE) ? Wself[n * FNODE + k]
                          : Wd[n * (FNODE + FEDGE) + (k - FNODE)];
    Wc[e] = (__bf16)v;
}

// ---------------------------------------------------------------------------
// Kernel 1: fused gather + sum + WMMA GEMM + bias -> x (fp32) in d_out.
// One block = 16 node rows x 128 out channels. 256 threads = 8 waves;
// wave w owns N-tile w (16 columns). A tile staged in LDS as bf16.
// ---------------------------------------------------------------------------
__global__ __launch_bounds__(256)
void gdc_fused_wmma(const float* __restrict__ node_repr,
                    const float* __restrict__ edge_repr,
                    const int* __restrict__ nn1, const int* __restrict__ ne1,
                    const int* __restrict__ nn2, const int* __restrict__ ne2,
                    const int* __restrict__ nn3, const int* __restrict__ ne3,
                    const int* __restrict__ nn4, const int* __restrict__ ne4,
                    const __bf16* __restrict__ Wc,
                    const float* __restrict__ bias,
                    float* __restrict__ x) {
    __shared__ __bf16 Alds[16][LDS_STRIDE];

    const int tile   = blockIdx.x;
    const int bucket = tile / TPB;            // 0..3 -> degree = bucket+1
    const int row0   = (tile % TPB) * 16;     // local row within bucket
    const int deg    = bucket + 1;

    const int* nn; const int* ne;
    switch (bucket) {
        case 0:  nn = nn1; ne = ne1; break;
        case 1:  nn = nn2; ne = ne2; break;
        case 2:  nn = nn3; ne = ne3; break;
        default: nn = nn4; ne = ne4; break;
    }

    // ---------------- Phase 1: build A[16][320] in LDS (all 256 threads) ----
    {
        const int r   = threadIdx.x >> 4;   // 0..15  row of tile
        const int sub = threadIdx.x & 15;   // 0..15  channel sub-lane
        const int li  = row0 + r;                       // local node in bucket
        const size_t self_gi = (size_t)bucket * NPD + li; // global node index

        // self features -> A[r][0..127]
        const float* sp = node_repr + self_gi * FNODE;
        #pragma unroll
        for (int k = 0; k < 8; ++k) {
            int c = sub + 16 * k;
            Alds[r][c] = (__bf16)sp[c];
        }

        // neighbor-node sum -> A[r][128..255]
        float an[8];
        #pragma unroll
        for (int k = 0; k < 8; ++k) an[k] = 0.f;
        for (int j = 0; j < deg; ++j) {
            const size_t gi = (size_t)nn[li * deg + j];
            const float* p = node_repr + gi * FNODE;
            #pragma unroll
            for (int k = 0; k < 8; ++k) an[k] += p[sub + 16 * k];
        }
        #pragma unroll
        for (int k = 0; k < 8; ++k)
            Alds[r][FNODE + sub + 16 * k] = (__bf16)an[k];

        // neighbor-edge sum -> A[r][256..319]
        float ae[4];
        #pragma unroll
        for (int k = 0; k < 4; ++k) ae[k] = 0.f;
        for (int j = 0; j < deg; ++j) {
            const size_t gi = (size_t)ne[li * deg + j];
            const float* p = edge_repr + gi * FEDGE;
            #pragma unroll
            for (int k = 0; k < 4; ++k) ae[k] += p[sub + 16 * k];
        }
        #pragma unroll
        for (int k = 0; k < 4; ++k)
            Alds[r][2 * FNODE + sub + 16 * k] = (__bf16)ae[k];
    }
    __syncthreads();

    // ---------------- Phase 2: per-wave 16x16 WMMA tile, K = 320 ------------
    const int wv   = threadIdx.x >> 5;        // 0..7 -> N-tile
    const int lane = threadIdx.x & 31;
    const int arow = lane & 15;               // A row / B column (mod 16)
    const int koff = (lane < 16) ? 0 : 8;     // K sub-offset per half-wave

    // B row pointer: Wc[bucket][n][.] with n = wv*16 + arow, k-major (320)
    const __bf16* Brow =
        Wc + ((size_t)(bucket * FOUT + wv * 16 + arow)) * KTOT;

    v8f acc = {};
    #pragma unroll
    for (int kc = 0; kc < KTOT / 32; ++kc) {
        const int k0 = kc * 32 + koff;
        // A fragment: lanes 0-15 hold K {0..7,16..23}, lanes 16-31 {8..15,24..31}
        v8bf alo = *(const v8bf*)&Alds[arow][k0];
        v8bf ahi = *(const v8bf*)&Alds[arow][k0 + 16];
        // B fragment: same K grouping, column n fixed per lane
        v8bf blo = *(const v8bf*)(Brow + k0);
        v8bf bhi = *(const v8bf*)(Brow + k0 + 16);
        v16bf a = __builtin_shufflevector(alo, ahi, 0, 1, 2, 3, 4, 5, 6, 7,
                                          8, 9, 10, 11, 12, 13, 14, 15);
        v16bf b = __builtin_shufflevector(blo, bhi, 0, 1, 2, 3, 4, 5, 6, 7,
                                          8, 9, 10, 11, 12, 13, 14, 15);
        acc = __builtin_amdgcn_wmma_f32_16x16x32_bf16(
            /*neg_a=*/false, a, /*neg_b=*/false, b,
            /*c_mod=*/(short)0, acc, /*reuse_a=*/false, /*reuse_b=*/false);
    }

    // ---------------- Phase 3: write x tile with bias -----------------------
    const int col = wv * 16 + (lane & 15);
    const float bv = bias[col];
    const int m0  = (lane < 16) ? 0 : 8;
    const size_t gbase = ((size_t)bucket * NPD + row0) * FOUT;
    #pragma unroll
    for (int v = 0; v < 8; ++v) {
        x[gbase + (size_t)(m0 + v) * FOUT + col] = acc[v] + bv;
    }
}

// ---------------------------------------------------------------------------
// Kernel 2: zero the 256-float stats buffer (sum[128], sumsq[128]).
// ---------------------------------------------------------------------------
__global__ void gdc_zero_stats(float* __restrict__ stats) {
    stats[threadIdx.x] = 0.f;
}

// ---------------------------------------------------------------------------
// Kernel 3: per-channel sum / sum-of-squares via f32 global atomics.
// ---------------------------------------------------------------------------
__global__ __launch_bounds__(256)
void gdc_stats(const float* __restrict__ x, float* __restrict__ stats) {
    const int c    = threadIdx.x & 127;
    const int half = threadIdx.x >> 7;      // 0 or 1 -> two rows per block
    float s = 0.f, sq = 0.f;
    for (long long row = (long long)blockIdx.x * 2 + half; row < NNODE;
         row += (long long)gridDim.x * 2) {
        float v = x[row * FOUT + c];
        s  += v;
        sq += v * v;
    }
    atomicAdd(&stats[c], s);
    atomicAdd(&stats[FOUT + c], sq);
}

// ---------------------------------------------------------------------------
// Kernel 4: BatchNorm (biased stats) + ReLU, in place on x.
// ---------------------------------------------------------------------------
__global__ __launch_bounds__(256)
void gdc_bn_relu(float* __restrict__ x, const float* __restrict__ stats) {
    const float invN = 1.0f / (float)NNODE;
    const size_t total = (size_t)NNODE * FOUT;
    for (size_t e = (size_t)blockIdx.x * blockDim.x + threadIdx.x; e < total;
         e += (size_t)gridDim.x * blockDim.x) {
        const int c = (int)(e & (FOUT - 1));
        const float mean = stats[c] * invN;
        const float var  = stats[FOUT + c] * invN - mean * mean;
        const float rstd = rsqrtf(var + 1e-5f);
        const float v = (x[e] - mean) * rstd;
        x[e] = v > 0.f ? v : 0.f;
    }
}

// ---------------------------------------------------------------------------
extern "C" void kernel_launch(void* const* d_in, const int* in_sizes, int n_in,
                              void* d_out, int out_size, void* d_ws, size_t ws_size,
                              hipStream_t stream) {
    const float* node_repr = (const float*)d_in[0];
    const float* edge_repr = (const float*)d_in[1];
    const int* nn1 = (const int*)d_in[2];
    const int* ne1 = (const int*)d_in[3];
    const int* nn2 = (const int*)d_in[4];
    const int* ne2 = (const int*)d_in[5];
    const int* nn3 = (const int*)d_in[6];
    const int* ne3 = (const int*)d_in[7];
    const int* nn4 = (const int*)d_in[8];
    const int* ne4 = (const int*)d_in[9];
    const float* W_self = (const float*)d_in[10];
    const float* W_d1   = (const float*)d_in[11];
    const float* W_d2   = (const float*)d_in[12];
    const float* W_d3   = (const float*)d_in[13];
    const float* W_d4   = (const float*)d_in[14];
    const float* bias   = (const float*)d_in[15];

    float* x = (float*)d_out;                       // [400000 x 128] fp32

    // Workspace layout: bf16 combined weights, then 256-float stats.
    __bf16* Wc   = (__bf16*)d_ws;                   // 4*128*320*2 = 327,680 B
    float* stats = (float*)((char*)d_ws + (size_t)4 * FOUT * KTOT * 2);

    // 0) combined bf16 weights
    {
        const int total = 4 * FOUT * KTOT;
        gdc_prep_weights<<<(total + 255) / 256, 256, 0, stream>>>(
            W_self, W_d1, W_d2, W_d3, W_d4, Wc);
    }
    // 1) zero BN stats
    gdc_zero_stats<<<1, 256, 0, stream>>>(stats);
    // 2) fused gather + WMMA -> x
    gdc_fused_wmma<<<NTILES, 256, 0, stream>>>(
        node_repr, edge_repr,
        nn1, ne1, nn2, ne2, nn3, ne3, nn4, ne4,
        Wc, bias, x);
    // 3) BN stats
    gdc_stats<<<1024, 256, 0, stream>>>(x, stats);
    // 4) normalize + ReLU in place
    gdc_bn_relu<<<2048, 256, 0, stream>>>(x, stats);
}
